// BiMamba2D_31971736552181
// MI455X (gfx1250) — matrix operand order repaired and checked by reading, not compiled, not run on hardware
//
#include <hip/hip_runtime.h>

typedef __attribute__((ext_vector_type(16))) __bf16 v16bf;
typedef __attribute__((ext_vector_type(8)))  __bf16 v8bf;
typedef __attribute__((ext_vector_type(8)))  float  v8f;

#define SEQS   384              // parallel sequences per branch (b*96)
#define LSEQ   96               // sequence length
#define DM     192              // d_model
#define DI     384              // d_inner
#define DSTATE 16
#define DTRANK 12
#define NROWS  (SEQS * LSEQ)    // 36864 rows for all GEMMs (multiple of 32)
#define XDBL_LD 48              // padded ld for (dt|B|C) = 44 -> 48
#define DTW_KPAD 32             // dt_proj K padded 12 -> 32 (zero weight tail)
#define XPW_NPAD 48             // x_proj NO padded 44 -> 48 (zero weight rows)

// ---------------------------------------------------------------------------
// f32 -> bf16 convert (plain weights)
// ---------------------------------------------------------------------------
__global__ void cvt_f32_bf16(const float* __restrict__ s, __bf16* __restrict__ d, int n) {
  int i = blockIdx.x * blockDim.x + threadIdx.x;
  if (i < n) d[i] = (__bf16)s[i];
}

// dt_proj weight: (4,384,12) -> (4,384,32) bf16, zero-padded in K
__global__ void pack_dtw(const float* __restrict__ s, __bf16* __restrict__ d, int total) {
  int i = blockIdx.x * blockDim.x + threadIdx.x;
  if (i >= total) return;
  int k = i % DTW_KPAD;
  int r = i / DTW_KPAD;        // dir*384 + row
  d[i] = (k < DTRANK) ? (__bf16)s[r * DTRANK + k] : (__bf16)0.0f;
}

// x_proj weight: (4,44,384) -> (4,48,384) bf16, zero-padded rows
__global__ void pack_xpw(const float* __restrict__ s, __bf16* __restrict__ d, int total) {
  int i = blockIdx.x * blockDim.x + threadIdx.x;
  if (i >= total) return;
  int k   = i % DI;
  int r   = (i / DI) % XPW_NPAD;
  int dir = i / (DI * XPW_NPAD);
  d[i] = (r < 44) ? (__bf16)s[((size_t)dir * 44 + r) * DI + k] : (__bf16)0.0f;
}

// ---------------------------------------------------------------------------
// Gather stage input into bf16 sequence-major layout [n=b*96+p][l][c].
// Both stages use: src[((b*96 + l')*96 + p)*192 + c], l' = flip ? 95-l : l.
// ---------------------------------------------------------------------------
__global__ void make_seq(const float* __restrict__ src, __bf16* __restrict__ dst,
                         int flip, int total) {
  int idx = blockIdx.x * blockDim.x + threadIdx.x;
  if (idx >= total) return;
  int c   = idx % DM;
  int row = idx / DM;          // n*96 + l
  int l   = row % LSEQ;
  int n   = row / LSEQ;        // b*96 + p
  int p   = n % 96;
  int b   = n / 96;
  int ls  = flip ? (LSEQ - 1 - l) : l;
  dst[idx] = (__bf16)src[(((b * 96 + ls) * 96 + p) * DM) + c];
}

// ---------------------------------------------------------------------------
// WMMA bf16 GEMM (unguarded fast path only):
//   C[M x (16*NTILES*gridDim.y)] = A[M x K] * W[NO x K]^T
// Preconditions (arranged by padding on the host side):
//   M % 32 == 0, K % 32 == 0, NO covered exactly by 16*NTILES*gridDim.y,
//   lda/ldw rows 16B-aligned at the 8-element k-chunks used below.
// One wave per 32 x (16*NTILES) tile.
// OUTMODE: 0 = f32 store, 1 = f32 accumulate, 2 = bf16 store.
// Fragment layouts per CDNA5 ISA 7.12.2 (16-bit A 16x32; B 32x16; f32 C/D 16x16).
// ---------------------------------------------------------------------------
template <int NTILES, int OUTMODE>
__global__ void gemm_wmma_bf16(const __bf16* __restrict__ A, int lda,
                               const __bf16* __restrict__ W, int ldw,
                               void* __restrict__ C, int ldc, int K) {
  const int lane = threadIdx.x & 31;
  const int row0 = blockIdx.x * 32;
  const int n0   = blockIdx.y * (16 * NTILES);

  const int mA   = lane & 15;      // A row within tile
  const int aSel = lane >> 4;      // half-wave select
  const int kbA  = aSel * 8;       // A k-base: 0 / 8 (+16 for hi chunk)
  const int nB   = lane & 15;      // B column within tile
  const int kbB  = aSel * 8;       // B k-base: 0 / 8 (+16 for hi chunk)

  v8f acc[2][NTILES];
#pragma unroll
  for (int i = 0; i < 2; i++)
#pragma unroll
    for (int j = 0; j < NTILES; j++)
#pragma unroll
      for (int r = 0; r < 8; r++) acc[i][j][r] = 0.0f;

  const __bf16* arow0 = A + (size_t)(row0 + mA) * lda + kbA;
  const __bf16* arow1 = arow0 + (size_t)16 * lda;
  const __bf16* wrow  = W + (size_t)(n0 + nB) * ldw + kbB;

  for (int k0 = 0; k0 < K; k0 += 32) {
    v16bf afrag[2];
#pragma unroll
    for (int i = 0; i < 2; i++) {
      const v8bf* p = (const v8bf*)((i ? arow1 : arow0) + k0);
      v8bf lo = p[0];
      v8bf hi = p[2];   // +16 elements
      v16bf f;
#pragma unroll
      for (int e = 0; e < 8; e++) { f[e] = lo[e]; f[8 + e] = hi[e]; }
      afrag[i] = f;
    }
#pragma unroll
    for (int j = 0; j < NTILES; j++) {
      const v8bf* p = (const v8bf*)(wrow + (size_t)(16 * j) * ldw + k0);
      v8bf lo = p[0];
      v8bf hi = p[2];
      v16bf b;
#pragma unroll
      for (int e = 0; e < 8; e++) { b[e] = lo[e]; b[8 + e] = hi[e]; }
      acc[0][j] = __builtin_amdgcn_wmma_f32_16x16x32_bf16(
          false, afrag[0], false, b, (short)0, acc[0][j], false, false);
      acc[1][j] = __builtin_amdgcn_wmma_f32_16x16x32_bf16(
          false, afrag[1], false, b, (short)0, acc[1][j], false, false);
    }
  }

#pragma unroll
  for (int i = 0; i < 2; i++)
#pragma unroll
    for (int j = 0; j < NTILES; j++)
#pragma unroll
      for (int r = 0; r < 8; r++) {
        int m = row0 + 16 * i + r + 8 * aSel;
        int n = n0 + 16 * j + nB;
        size_t idx = (size_t)m * ldc + n;
        float v = acc[i][j][r];
        if (OUTMODE == 2)      ((__bf16*)C)[idx] = (__bf16)v;
        else if (OUTMODE == 1) ((float*)C)[idx] += v;
        else                   ((float*)C)[idx]  = v;
      }
}

// ---------------------------------------------------------------------------
// Causal depthwise conv1d (k=3, pad-left 2) + bias + SiLU.
// Input: xz bf16 [NROWS x 768] (xs = cols 0..383). Output: xsc bf16 [NROWS x 384].
// ---------------------------------------------------------------------------
__global__ void conv_silu(const __bf16* __restrict__ xz,
                          const float* __restrict__ cw, const float* __restrict__ cb,
                          __bf16* __restrict__ xsc, int total) {
  int idx = blockIdx.x * blockDim.x + threadIdx.x;
  if (idx >= total) return;
  int d   = idx % DI;
  int row = idx / DI;          // n*96 + l
  int l   = row % LSEQ;
  float w0 = cw[d * 3 + 0], w1 = cw[d * 3 + 1], w2 = cw[d * 3 + 2];
  float acc = cb[d];
  float xm2 = (l >= 2) ? (float)xz[(size_t)(row - 2) * 768 + d] : 0.0f;
  float xm1 = (l >= 1) ? (float)xz[(size_t)(row - 1) * 768 + d] : 0.0f;
  float x0  = (float)xz[(size_t)row * 768 + d];
  acc += xm2 * w0 + xm1 * w1 + x0 * w2;
  float s = acc / (1.0f + __expf(-acc));
  xsc[(size_t)row * DI + d] = (__bf16)s;
}

// ---------------------------------------------------------------------------
// Selective scan, fused softplus(dt) + D-skip + silu(z) gating.
// One thread per (sequence, channel); 16-wide state in registers, 96 steps.
// ---------------------------------------------------------------------------
__global__ void selective_scan(const __bf16* __restrict__ xsc,
                               const __bf16* __restrict__ dtraw,
                               const __bf16* __restrict__ xdbl,
                               const __bf16* __restrict__ xz,   // z at col 384
                               const float* __restrict__ dtb,
                               const float* __restrict__ Alog,
                               const float* __restrict__ Dsk,
                               __bf16* __restrict__ yg) {
  int tid = blockIdx.x * blockDim.x + threadIdx.x;
  if (tid >= SEQS * DI) return;
  int d = tid % DI;
  int n = tid / DI;

  float Av[DSTATE];
#pragma unroll
  for (int s = 0; s < DSTATE; s++) Av[s] = -__expf(Alog[d * DSTATE + s]);
  float Dv = Dsk[d];
  float bias = dtb[d];

  float h[DSTATE];
#pragma unroll
  for (int s = 0; s < DSTATE; s++) h[s] = 0.0f;

  int rowbase = n * LSEQ;
  for (int l = 0; l < LSEQ; l++) {
    int row = rowbase + l;
    float u  = (float)xsc[(size_t)row * DI + d];
    float xr = (float)dtraw[(size_t)row * DI + d] + bias;
    float dt = (xr > 20.0f) ? xr : __logf(1.0f + __expf(xr));
    float dtu = dt * u;
    float y = 0.0f;
#pragma unroll
    for (int s = 0; s < DSTATE; s++) {
      float Bv = (float)xdbl[(size_t)row * XDBL_LD + DTRANK + s];
      float Cv = (float)xdbl[(size_t)row * XDBL_LD + DTRANK + DSTATE + s];
      h[s] = h[s] * __expf(dt * Av[s]) + dtu * Bv;
      y += h[s] * Cv;
    }
    y += u * Dv;
    float z = (float)xz[(size_t)row * 768 + DI + d];
    float g = z / (1.0f + __expf(-z));
    yg[(size_t)row * DI + d] = (__bf16)(y * g);
  }
}

// ---------------------------------------------------------------------------
extern "C" void kernel_launch(void* const* d_in, const int* in_sizes, int n_in,
                              void* d_out, int out_size, void* d_ws, size_t ws_size,
                              hipStream_t stream) {
  (void)in_sizes; (void)n_in; (void)out_size; (void)ws_size;

  const float* x    = (const float*)d_in[0];
  const float* ipw  = (const float*)d_in[1];   // (4, 768, 192)
  const float* cw   = (const float*)d_in[2];   // (4, 384, 1, 3)
  const float* cb   = (const float*)d_in[3];   // (4, 384)
  const float* xpw  = (const float*)d_in[4];   // (4, 44, 384)
  const float* dtw  = (const float*)d_in[5];   // (4, 384, 12)
  const float* dtb  = (const float*)d_in[6];   // (4, 384)
  const float* Alog = (const float*)d_in[7];   // (4, 384, 16)
  const float* Dsk  = (const float*)d_in[8];   // (4, 384)
  const float* opw  = (const float*)d_in[9];   // (4, 192, 384)
  float* out = (float*)d_out;

  char* base = (char*)d_ws;
  size_t off = 0;
  auto take = [&](size_t elems, size_t esz) -> char* {
    char* p = base + off;
    off += ((elems * esz + 255) & ~(size_t)255);
    return p;
  };
  __bf16* ipw_bf   = (__bf16*)take((size_t)4 * 768 * 192, 2);
  __bf16* xpw_bf   = (__bf16*)take((size_t)4 * XPW_NPAD * DI, 2);   // padded rows
  __bf16* dtw_bf   = (__bf16*)take((size_t)4 * DI * DTW_KPAD, 2);   // padded K
  __bf16* opw_bf   = (__bf16*)take((size_t)4 * 192 * 384, 2);
  __bf16* seq_bf   = (__bf16*)take((size_t)NROWS * DM, 2);
  __bf16* xz_bf    = (__bf16*)take((size_t)NROWS * 768, 2);
  __bf16* xsc_bf   = (__bf16*)take((size_t)NROWS * DI, 2);
  __bf16* xdbl_bf  = (__bf16*)take((size_t)NROWS * XDBL_LD, 2);
  __bf16* dtraw_bf = (__bf16*)take((size_t)NROWS * DI, 2);
  __bf16* yg_bf    = (__bf16*)take((size_t)NROWS * DI, 2);
  float*  s1       = (float*) take((size_t)NROWS * DM, 4);

  const int TPB = 256;
  auto blocks = [&](int n) { return dim3((unsigned)((n + TPB - 1) / TPB)); };

  // weights -> bf16 (with zero padding where needed)
  cvt_f32_bf16<<<blocks(4 * 768 * 192), TPB, 0, stream>>>(ipw, ipw_bf, 4 * 768 * 192);
  pack_xpw<<<blocks(4 * XPW_NPAD * DI), TPB, 0, stream>>>(xpw, xpw_bf, 4 * XPW_NPAD * DI);
  pack_dtw<<<blocks(4 * DI * DTW_KPAD), TPB, 0, stream>>>(dtw, dtw_bf, 4 * DI * DTW_KPAD);
  cvt_f32_bf16<<<blocks(4 * 192 * 384), TPB, 0, stream>>>(opw, opw_bf, 4 * 192 * 384);

  auto run_branch = [&](int dir, const float* seqsrc, int flip, float* outC, int accum) {
    // 1) layout/flip gather -> bf16
    make_seq<<<blocks(NROWS * DM), TPB, 0, stream>>>(seqsrc, seq_bf, flip, NROWS * DM);
    // 2) in_proj: [36864 x 192] x [768 x 192]^T -> xz bf16 [36864 x 768]
    gemm_wmma_bf16<4, 2><<<dim3(NROWS / 32, 768 / 64), 32, 0, stream>>>(
        seq_bf, DM, ipw_bf + (size_t)dir * 768 * 192, DM, xz_bf, 768, DM);
    // 3) causal depthwise conv + SiLU -> xsc bf16
    conv_silu<<<blocks(NROWS * DI), TPB, 0, stream>>>(
        xz_bf, cw + (size_t)dir * DI * 3, cb + (size_t)dir * DI, xsc_bf, NROWS * DI);
    // 4) x_proj: K=384, NO=48 (rows 44..47 zero weights) -> xdbl bf16 (ld 48)
    gemm_wmma_bf16<3, 2><<<dim3(NROWS / 32, 1), 32, 0, stream>>>(
        xsc_bf, DI, xpw_bf + (size_t)dir * XPW_NPAD * DI, DI, xdbl_bf, XDBL_LD, DI);
    // 5) dt_proj: K=32 (cols 12..31 of W are zero), NO=384 -> dtraw bf16
    gemm_wmma_bf16<4, 2><<<dim3(NROWS / 32, 384 / 64), 32, 0, stream>>>(
        xdbl_bf, XDBL_LD, dtw_bf + (size_t)dir * DI * DTW_KPAD, DTW_KPAD,
        dtraw_bf, DI, DTW_KPAD);
    // 6) selective scan + softplus + D-skip + silu(z) gating -> yg bf16
    selective_scan<<<blocks(SEQS * DI), TPB, 0, stream>>>(
        xsc_bf, dtraw_bf, xdbl_bf, xz_bf,
        dtb + (size_t)dir * DI, Alog + (size_t)dir * DI * DSTATE,
        Dsk + (size_t)dir * DI, yg_bf);
    // 7) out_proj: K=384, NO=192, write or accumulate f32
    if (accum)
      gemm_wmma_bf16<4, 1><<<dim3(NROWS / 32, 192 / 64), 32, 0, stream>>>(
          yg_bf, DI, opw_bf + (size_t)dir * 192 * 384, DI, outC, DM, DI);
    else
      gemm_wmma_bf16<4, 0><<<dim3(NROWS / 32, 192 / 64), 32, 0, stream>>>(
          yg_bf, DI, opw_bf + (size_t)dir * 192 * 384, DI, outC, DM, DI);
  };

  // stage 1 (w-direction): params 0 (fwd) + 1 (rev) -> s1
  run_branch(0, x, 0, s1, 0);
  run_branch(1, x, 1, s1, 1);
  // stage 2 (h-direction): params 2 (fwd) + 3 (rev) -> d_out
  // stage-2 row index (b*96+h)*96+w == d_out flat layout, so write directly.
  run_branch(2, s1, 0, out, 0);
  run_branch(3, s1, 1, out, 1);
}
